// ManiBlock_25013889532078
// MI455X (gfx1250) — matrix operand order repaired and checked by reading, not compiled, not run on hardware
//
#include <hip/hip_runtime.h>

typedef __attribute__((ext_vector_type(16))) _Float16 v16h;
typedef __attribute__((ext_vector_type(8)))  float    v8f;
typedef __attribute__((ext_vector_type(4)))  float    v4f;
typedef unsigned int u32x4 __attribute__((ext_vector_type(4)));
typedef int          i32x8 __attribute__((ext_vector_type(8)));
typedef int          i32x4 __attribute__((ext_vector_type(4)));

#define CH    4
#define KN    16
#define IN_D  400
#define OUT_D 100
#define PPK   10      // P
#define NCOL  10
#define NPAD  16
#define MPAD  112     // 7 * 16
#define HROW  50      // pooled rows
#define KCHUNKS 13    // ceil(400/32)

__device__ __forceinline__ float wave_sum(float v) {
#pragma unroll
  for (int m = 16; m >= 1; m >>= 1) v += __shfl_xor(v, m, 32);
  return v;
}

__global__ __launch_bounds__(256)
void maniblock_kernel(const float* __restrict__ x,
                      const float* __restrict__ w,
                      float* __restrict__ out) {
  // TDM staging target: raw x[b,c] tile (400x10 f32, contiguous)
  __shared__ __align__(16) float sX[IN_D * NCOL];                 // 16000 B
  // B operands pre-swizzled into WMMA fragment layout:
// halfword index = ((kt*32 + lane) * 16 + e); K = 32kt + 8*(lane>>4) + (e&7) + 16*(e>>3), N = lane&15
  __shared__ __align__(32) _Float16 sBfrag[KCHUNKS * 32 * 16];    // 13312 B
  __shared__ float sR[MPAD * NPAD];                               // 7168 B
  __shared__ float sQp[HROW * PPK];
  __shared__ float sN[HROW * PPK];

  const int ck   = blockIdx.x;           // c*16 + k
  const int b    = blockIdx.y;
  const int c    = ck >> 4;
  const int kk   = ck & 15;
  const int tid  = threadIdx.x;
  const int lane = tid & 31;
  const int wid  = tid >> 5;

  const float* xBase = x + (size_t)(b * CH + c) * IN_D * NCOL;

  // ---- Stage 1a: Tensor Data Mover DMA of x[b,c] (4000 f32) into LDS ----
  if (wid == 0) {
    unsigned long long gaddr = (unsigned long long)(const void*)xBase;
    unsigned int lds_off = (unsigned int)(unsigned long long)(void*)&sX[0];

    u32x4 g0;
    g0[0] = 1u;                                           // count=1 (valid user D#)
    g0[1] = lds_off;                                      // lds_addr
    g0[2] = (unsigned int)(gaddr & 0xFFFFFFFFull);        // global_addr[31:0]
    g0[3] = (unsigned int)((gaddr >> 32) & 0x01FFFFFFull) // global_addr[56:32]
            | (2u << 30);                                 // type=2 ("image")

    const unsigned int NELEM = IN_D * NCOL;               // 4000 elements of 4B
    i32x8 g1 = {};
    g1[0] = (int)(2u << 16);                              // workgroup_mask=0, data_size=2 (4B)
    g1[1] = (int)((NELEM & 0xFFFFu) << 16);               // tensor_dim0[15:0]
    g1[2] = (int)(((NELEM >> 16) & 0xFFFFu) | (1u << 16));// tensor_dim0[31:16] | tensor_dim1[15:0]=1
    g1[3] = (int)((NELEM & 0xFFFFu) << 16);               // tensor_dim1 hi=0 | tile_dim0=4000
    g1[4] = (int)(1u);                                    // tile_dim1=1, tile_dim2=0
    g1[5] = (int)NELEM;                                   // tensor_dim0_stride[31:0]
    g1[6] = 0;                                            // stride hi | dim1_stride lo
    g1[7] = 0;

    i32x4 gz = {};
#if defined(__clang_major__) && (__clang_major__ >= 23)
    i32x8 gz8 = {};
    __builtin_amdgcn_tensor_load_to_lds(g0, g1, gz, gz, gz8, 0);
#else
    __builtin_amdgcn_tensor_load_to_lds(g0, g1, gz, gz, 0);
#endif
    __builtin_amdgcn_s_wait_tensorcnt((short)0);
  }
  __syncthreads();

  // ---- Stage 1b: swizzle sX (f32) -> sBfrag (f16, WMMA B layout, zero padded) ----
  for (int idx = tid; idx < KCHUNKS * 32 * 16; idx += 256) {
    int kt = idx >> 9;                 // /512
    int rem = idx & 511;
    int ln = rem >> 4;
    int e  = rem & 15;
    int K = 32 * kt + 8 * (ln >> 4) + (e & 7) + 16 * (e >> 3);
    int N = ln & 15;
    float v = (K < IN_D && N < NCOL) ? sX[K * NCOL + N] : 0.0f;
    sBfrag[idx] = (_Float16)v;
  }
  __syncthreads();

  // ---- Stage 2: r = w[c,k] @ x[b,c] via v_wmma_f32_16x16x32_f16 ----
  if (wid < 7) {
    const int half = lane >> 4;
    const int o    = wid * 16 + (lane & 15);            // output row (M)
    const int oc   = (o < OUT_D) ? o : (OUT_D - 1);     // clamp; rows >=100 never read back
    const float* wRow = w + (size_t)((c * KN + kk) * OUT_D + oc) * IN_D;
    const _Float16* fragBase = sBfrag + (size_t)lane * 16;

    v8f acc = {};
#pragma unroll 2
    for (int kt = 0; kt < 12; ++kt) {
      const int i0 = kt * 32 + 8 * half;
      v4f a0 = *(const v4f*)(wRow + i0);
      v4f a1 = *(const v4f*)(wRow + i0 + 4);
      v4f a2 = *(const v4f*)(wRow + i0 + 16);
      v4f a3 = *(const v4f*)(wRow + i0 + 20);
      __builtin_prefetch(wRow + i0 + 64, 0, 1);         // global_prefetch_b8

      v16h A;
#pragma unroll
      for (int e = 0; e < 4; ++e) {
        A[e]      = (_Float16)a0[e];
        A[e + 4]  = (_Float16)a1[e];
        A[e + 8]  = (_Float16)a2[e];
        A[e + 12] = (_Float16)a3[e];
      }
      v16h B = *(const v16h*)(fragBase + kt * 512);
      acc = __builtin_amdgcn_wmma_f32_16x16x32_f16(
          false, A, false, B, (short)0, acc, false, false);
    }
    { // epilogue kt=12: K=384..415, only 384..399 valid
      const int i0 = 384 + 8 * half;
      v4f a0 = *(const v4f*)(wRow + i0);
      v4f a1 = *(const v4f*)(wRow + i0 + 4);
      v16h A;
#pragma unroll
      for (int e = 0; e < 4; ++e) {
        A[e]      = (_Float16)a0[e];
        A[e + 4]  = (_Float16)a1[e];
        A[e + 8]  = (_Float16)0.0f;
        A[e + 12] = (_Float16)0.0f;
      }
      v16h B = *(const v16h*)(fragBase + 12 * 512);
      acc = __builtin_amdgcn_wmma_f32_16x16x32_f16(
          false, A, false, B, (short)0, acc, false, false);
    }

    // C/D layout: element r -> row (r + 8*half), col = lane&15
#pragma unroll
    for (int r = 0; r < 8; ++r)
      sR[(wid * 16 + r + 8 * half) * NPAD + (lane & 15)] = acc[r];
  }
  __syncthreads();

  // ---- Stage 3 (wave 0): thin QR (MGS) of r (100x10); pool to Qp (50x10); ----
  // ---- then orthonormal null-space basis of Qp == eigh(Pp)[..., :10]. ----
  if (wid == 0) {
    for (int j = 0; j < PPK; ++j) {
      float s = 0.0f;
      for (int i = lane; i < OUT_D; i += 32) { float v = sR[i * NPAD + j]; s += v * v; }
      s = wave_sum(s);
      float inv = rsqrtf(s + 1e-30f);
      for (int i = lane; i < OUT_D; i += 32) sR[i * NPAD + j] *= inv;
      for (int l = j + 1; l < PPK; ++l) {
        float d = 0.0f;
        for (int i = lane; i < OUT_D; i += 32) d += sR[i * NPAD + j] * sR[i * NPAD + l];
        d = wave_sum(d);
        for (int i = lane; i < OUT_D; i += 32) sR[i * NPAD + l] -= d * sR[i * NPAD + j];
      }
    }

    for (int e = lane; e < HROW * PPK; e += 32) {
      int i = e / PPK, j = e % PPK;
      sQp[e] = 0.5f * (sR[(2 * i) * NPAD + j] + sR[(2 * i + 1) * NPAD + j]);
      sN[e] = 0.0f;
    }

    int cnt = 0;
    for (int cand = 0; cand < HROW && cnt < PPK; ++cand) {
      const int i0r = lane;
      const int i1r = lane + 32;
      const int i1c = (i1r < HROW) ? i1r : 0;
      const float m1 = (i1r < HROW) ? 1.0f : 0.0f;

      float v0 = (i0r == cand) ? 1.0f : 0.0f;
      float v1 = (i1r == cand) ? 1.0f : 0.0f;
#pragma unroll
      for (int j = 0; j < PPK; ++j) {
        float a = sQp[cand * PPK + j];
        v0 -= a * sQp[i0r * PPK + j];
        v1 -= a * sQp[i1c * PPK + j];
      }
      for (int t = 0; t < cnt; ++t) {
        float bc = sN[cand * PPK + t];
        v0 -= bc * sN[i0r * PPK + t];
        v1 -= bc * sN[i1c * PPK + t];
      }
      v1 *= m1;

      float n2 = wave_sum(v0 * v0 + v1 * v1);
      if (n2 > 1e-6f) {
        float inv = rsqrtf(n2);
        sN[i0r * PPK + cnt] = v0 * inv;
        if (i1r < HROW) sN[i1r * PPK + cnt] = v1 * inv;
        ++cnt;
      }
    }
  }
  __syncthreads();

  // ---- Stage 4: write U[b, ck, :, :10]  (64,64,50,10 row-major) ----
  float* outBase = out + (size_t)(b * 64 + ck) * HROW * PPK;
  for (int e = tid; e < HROW * PPK; e += 256) outBase[e] = sN[e];
}

extern "C" void kernel_launch(void* const* d_in, const int* in_sizes, int n_in,
                              void* d_out, int out_size, void* d_ws, size_t ws_size,
                              hipStream_t stream) {
  (void)in_sizes; (void)n_in; (void)out_size; (void)d_ws; (void)ws_size;
  const float* x = (const float*)d_in[0];   // (64, 4, 400, 10) f32
  const float* w = (const float*)d_in[1];   // (4, 16, 100, 400) f32
  float* out = (float*)d_out;               // (64, 64, 50, 10) f32

  dim3 grid(64, 64);
  maniblock_kernel<<<grid, 256, 0, stream>>>(x, w, out);
}
